// StackRNN_45260365365276
// MI455X (gfx1250) — matrix operand order repaired and checked by reading, not compiled, not run on hardware
//
#include <hip/hip_runtime.h>
#include <hip/hip_bf16.h>

typedef _Float16 h8   __attribute__((ext_vector_type(8)));
typedef _Float16 v16h __attribute__((ext_vector_type(16)));
typedef float    v8f  __attribute__((ext_vector_type(8)));

// Padded activation layout: per batch, 1032 rows of 512 halves;
// rows 0..7 are zero (causal pad), rows 8..1031 hold t=0..1023.
#define TP      1032
#define ROWSZ   512
#define NBATCH  32
#define TLEN    1024
#define ACT_ELEMS ((long)NBATCH * TP * ROWSZ)     // 16,908,288 halves (~33.8 MB)
#define WL_ELEMS  (512l * 4096)                   // per-layer packed weights (1<<21)

// ---------------------------------------------------------------------------
// Zero the 8 pad rows of each batch in both ping-pong activation buffers.
// ---------------------------------------------------------------------------
__global__ void zero_pads_kernel(_Float16* __restrict__ a,
                                 _Float16* __restrict__ b) {
    int i = blockIdx.x * blockDim.x + threadIdx.x;   // < 131072
    int bat = i >> 12;                                // 8*512 = 4096 per batch
    int rem = i & 4095;
    long idx = ((long)bat * TP) * ROWSZ + rem;        // first 8 rows of batch
    a[idx] = (_Float16)0.0f;
    b[idx] = (_Float16)0.0f;
}

// ---------------------------------------------------------------------------
// x (f32, [b][t][k]) -> padded f16 activations
// ---------------------------------------------------------------------------
__global__ void convert_x_kernel(const float* __restrict__ x,
                                 _Float16* __restrict__ dst, int n) {
    int i = blockIdx.x * blockDim.x + threadIdx.x;
    if (i >= n) return;
    int bat = i >> 19;                 // 1024*512 = 2^19 per batch
    int rem = i & 524287;
    long di = ((long)bat * TP + 8) * ROWSZ + rem;
    dst[di] = (_Float16)x[i];
}

// ---------------------------------------------------------------------------
// W (f32, [l][o][h][k]) -> f16 packed in B-fragment order:
//   wpack[l][tile(32)][kstep(128)][lane(32)][e(16)]
//   lane = g*16 + r : column o = tile*16 + r, K = kstep*32 + g*8 + e (+8 if e >= 8)
// ---------------------------------------------------------------------------
__global__ void pack_w_kernel(const float* __restrict__ W,
                              _Float16* __restrict__ dst, int n) {
    int d = blockIdx.x * blockDim.x + threadIdx.x;
    if (d >= n) return;
    int l    = d >> 21;
    int rem  = d & ((1 << 21) - 1);
    int e    = rem & 15;
    int lane = (rem >> 4) & 31;
    int ks   = (rem >> 9) & 127;
    int tile = rem >> 16;              // 0..31
    int r = lane & 15, g = lane >> 4;
    int o  = tile * 16 + r;
    int kk = ks * 32 + g * 8 + e + ((e >> 3) << 3);
    dst[d] = (_Float16)W[((long)l << 21) + (long)o * 4096 + kk];
}

// ---------------------------------------------------------------------------
__device__ __forceinline__ v16h load_a(const _Float16* p) {
    // A frag: elems 0..7 at p, elems 8..15 at p+16 (K = g*8.. and 16+g*8..)
    h8 lo = *(const h8*)(p);
    h8 hi = *(const h8*)(p + 16);
    return __builtin_shufflevector(lo, hi, 0,1,2,3,4,5,6,7,8,9,10,11,12,13,14,15);
}
__device__ __forceinline__ v16h load_b(const _Float16* p) {
    // Packed B frag: 32 contiguous bytes per lane
    h8 lo = *(const h8*)(p);
    h8 hi = *(const h8*)(p + 8);
    return __builtin_shufflevector(lo, hi, 0,1,2,3,4,5,6,7,8,9,10,11,12,13,14,15);
}
#define WMMA(A, B, C) __builtin_amdgcn_wmma_f32_16x16x32_f16( \
        false, (A), false, (B), (short)0, (C), false, false)

// ---------------------------------------------------------------------------
// One layer. Grid: 1024 blocks = 128 m-blocks x 8 n-groups.
// Block covers 256 rows x 64 cols; wave wv covers rows [wv*32, wv*32+32).
// All 8 waves share the SAME 4 B-column tiles -> identical B addresses
// across waves hit in WGP$/L0 (one miss per block), A is wave-private.
// Wave tile: 32x64 = 2 A-frags x 4 B-frags -> 8 WMMA per K-step, 128 steps,
// register double-buffered so next-step loads overlap current WMMAs.
// ---------------------------------------------------------------------------
__global__ __launch_bounds__(256)
void stackrnn_layer_kernel(const _Float16* __restrict__ actIn,
                           const _Float16* __restrict__ wl,    // packed, this layer
                           const float*    __restrict__ bias,  // this layer
                           _Float16* __restrict__ actOut,
                           float* __restrict__ finalOut,
                           int isLast)
{
    const int lane = threadIdx.x & 31;
    const int wv   = threadIdx.x >> 5;       // 0..7
    const int r    = lane & 15;
    const int g    = lane >> 4;

    const int n0   = (blockIdx.x & 7) << 6;            // 64 cols, shared by block
    const int bt0  = ((blockIdx.x >> 3) << 8) + (wv << 5); // 32 rows per wave
    const int bat  = bt0 >> 10;
    const int t0   = bt0 & 1023;
    const int rowp = bat * TP + 8 + t0;      // padded row of wave's first row

    // A bases: padded row (rowp + mt*16 + r - 7), always in-bounds (pads = 0)
    const _Float16* aB0 = actIn + ((long)(rowp + r - 7) << 9) + (g << 3);
    const _Float16* aB1 = aB0 + (16l << 9);
    // B base: packed fragments; +512 halves per kstep, +65536 per n-tile
    const _Float16* bp  = wl + (((long)(n0 >> 4) * 128) * 32 + lane) * 16;

    v8f acc[2][4] = {};

    v16h a0 = load_a(aB0);
    v16h a1 = load_a(aB1);
    v16h b0 = load_b(bp);
    v16h b1 = load_b(bp + 65536);
    v16h b2 = load_b(bp + 131072);
    v16h b3 = load_b(bp + 196608);

    for (int ks = 1; ks < 128; ++ks) {
        const long ao = (long)ks * 32;
        const long bo = (long)ks * 512;
        v16h na0 = load_a(aB0 + ao);
        v16h na1 = load_a(aB1 + ao);
        v16h nb0 = load_b(bp + bo);
        v16h nb1 = load_b(bp + 65536 + bo);
        v16h nb2 = load_b(bp + 131072 + bo);
        v16h nb3 = load_b(bp + 196608 + bo);

        acc[0][0] = WMMA(a0, b0, acc[0][0]);
        acc[0][1] = WMMA(a0, b1, acc[0][1]);
        acc[0][2] = WMMA(a0, b2, acc[0][2]);
        acc[0][3] = WMMA(a0, b3, acc[0][3]);
        acc[1][0] = WMMA(a1, b0, acc[1][0]);
        acc[1][1] = WMMA(a1, b1, acc[1][1]);
        acc[1][2] = WMMA(a1, b2, acc[1][2]);
        acc[1][3] = WMMA(a1, b3, acc[1][3]);

        a0 = na0; a1 = na1;
        b0 = nb0; b1 = nb1; b2 = nb2; b3 = nb3;
    }
    // drain
    acc[0][0] = WMMA(a0, b0, acc[0][0]);
    acc[0][1] = WMMA(a0, b1, acc[0][1]);
    acc[0][2] = WMMA(a0, b2, acc[0][2]);
    acc[0][3] = WMMA(a0, b3, acc[0][3]);
    acc[1][0] = WMMA(a1, b0, acc[1][0]);
    acc[1][1] = WMMA(a1, b1, acc[1][1]);
    acc[1][2] = WMMA(a1, b2, acc[1][2]);
    acc[1][3] = WMMA(a1, b3, acc[1][3]);

    // Epilogue: D element j of lane (r,g) -> (row = j + 8g, col = r)
#pragma unroll
    for (int mt = 0; mt < 2; ++mt) {
#pragma unroll
        for (int i = 0; i < 4; ++i) {
            const int o  = n0 + (i << 4) + r;
            const float bv = bias[o];
#pragma unroll
            for (int j = 0; j < 8; ++j) {
                const int mrow = (mt << 4) + (g << 3) + j;     // 0..31 in tile
                const float v = tanhf(acc[mt][i][j] + bv);
                actOut[((long)(rowp + mrow) << 9) + o] = (_Float16)v;
                if (isLast) finalOut[((long)(bt0 + mrow) << 9) + o] = v;
            }
        }
    }
}

// ---------------------------------------------------------------------------
extern "C" void kernel_launch(void* const* d_in, const int* in_sizes, int n_in,
                              void* d_out, int out_size, void* d_ws, size_t ws_size,
                              hipStream_t stream) {
    const float* x = (const float*)d_in[0];   // (32,1024,512)
    const float* W = (const float*)d_in[1];   // (4,512,8,512)
    const float* b = (const float*)d_in[2];   // (4,512)
    float* out = (float*)d_out;

    _Float16* actA = (_Float16*)d_ws;          // ~33.8 MB
    _Float16* actB = actA + ACT_ELEMS;         // ~33.8 MB
    _Float16* w16  = actB + ACT_ELEMS;         // 16 MB packed weights

    const int NPAD = NBATCH * 8 * ROWSZ;       // 131072
    const int NX   = NBATCH * TLEN * ROWSZ;    // 16,777,216
    const int NW   = 4 << 21;                  // 8,388,608

    zero_pads_kernel<<<(NPAD + 255) / 256, 256, 0, stream>>>(actA, actB);
    convert_x_kernel<<<(NX + 255) / 256, 256, 0, stream>>>(x, actA, NX);
    pack_w_kernel<<<(NW + 255) / 256, 256, 0, stream>>>(W, w16, NW);

    _Float16* cur = actA;
    _Float16* nxt = actB;
    for (int l = 0; l < 4; ++l) {
        stackrnn_layer_kernel<<<1024, 256, 0, stream>>>(
            cur, w16 + (long)l * WL_ELEMS, b + (long)l * 512,
            nxt, out, (l == 3) ? 1 : 0);
        _Float16* t = cur; cur = nxt; nxt = t;
    }
}
